// MultiScaleAttention_67886253080866
// MI455X (gfx1250) — matrix operand order repaired and checked by reading, not compiled, not run on hardware
//
#include <hip/hip_runtime.h>
#include <hip/hip_bf16.h>
#include <cstdint>
#include <cstddef>

// ---------------- problem constants ----------------
constexpr int T_ = 8, H_ = 28, W_ = 28, DIM = 768, NH = 12, HD = 64;
constexpr int SP  = T_ * H_ * W_;      // 6272 spatial q tokens
constexpr int N_  = 1 + SP;            // 6273 q tokens (cls + spatial)
constexpr int KT = 8, KH = 14, KW = 14;
constexpr int KSP = KT * KH * KW;      // 1568 pooled spatial tokens
constexpr int KN  = 1 + KSP;           // 1569 kv tokens
constexpr int KNP = 1600;              // kv tokens padded to multiple of 32
constexpr int QT  = (N_ + 15) / 16;    // 393 q tiles of 16 rows
constexpr int NP  = QT * 16;           // 6288 padded q tokens
constexpr float SCALE_ = 0.125f;       // 64^-0.5

typedef __attribute__((ext_vector_type(16))) __bf16 v16bf;
typedef __attribute__((ext_vector_type(8)))  __bf16 v8bf;
typedef __attribute__((ext_vector_type(8)))  float  v8f;

#define CAT16(a, b) __builtin_shufflevector(a, b, 0,1,2,3,4,5,6,7,8,9,10,11,12,13,14,15)

// ---------------- fp32 -> bf16 convert ----------------
__global__ void k_f32_to_bf16(const float* __restrict__ in, __bf16* __restrict__ out, size_t n) {
    size_t i = (size_t)blockIdx.x * blockDim.x + threadIdx.x;
    if (i < n) out[i] = (__bf16)in[i];
}

// ---------------- WMMA bf16 GEMM: C[M][Nout] = A[M][K] * W[Nout][K]^T + bias ----------------
// one 16x64 output tile per wave (1 A fragment reused over 4 B fragments), K stepped by 32.
// All fragment loads are unconditional b128; A row pointer is clamped (garbage rows are
// only ever accumulated into output rows that the epilogue never stores).
__global__ void k_gemm_bf16(const __bf16* __restrict__ A, const __bf16* __restrict__ Wt,
                            const float* __restrict__ bias, float* __restrict__ C,
                            int M, int K, int Nout) {
    int wave   = (int)((blockIdx.x * (size_t)blockDim.x + threadIdx.x) >> 5);
    int lane   = threadIdx.x & 31;
    int tilesN = Nout >> 6;                         // 64-wide N tiles
    int tilesM = (M + 15) >> 4;
    if (wave >= tilesM * tilesN) return;
    int tm = wave / tilesN, tn = wave % tilesN;
    int lo = lane & 15, hi = lane >> 4;

    int rowA = tm * 16 + lo;
    if (rowA >= M) rowA = M - 1;                    // clamp: no predication on loads
    const __bf16* arow  = A  + (size_t)rowA * K;
    const __bf16* wbase = Wt + (size_t)(tn * 64 + lo) * K;

    v8f acc[4] = {};
    for (int kk = 0; kk < K; kk += 32) {
        // A fragment 16x32: lane row rowA, K chunks {hi*8..+7, 16+hi*8..+7}
        const __bf16* ap = arow + kk + hi * 8;
        v8bf a0 = *(const v8bf*)ap;
        v8bf a1 = *(const v8bf*)(ap + 16);
        v16bf a = CAT16(a0, a1);
        if (kk + 32 < K) __builtin_prefetch(ap + 32, 0, 0);
        #pragma unroll
        for (int s = 0; s < 4; ++s) {
            // B fragment 32x16: lane column (tn*64 + s*16 + lo), K chunks {hi*8..+7, 16+hi*8..+7}
            const __bf16* bp = wbase + (size_t)s * 16 * K + kk + hi * 8;
            v8bf b0 = *(const v8bf*)bp;
            v8bf b1 = *(const v8bf*)(bp + 16);
            v16bf b = CAT16(b0, b1);
            acc[s] = __builtin_amdgcn_wmma_f32_16x16x32_bf16(false, a, false, b, (short)0, acc[s], false, false);
        }
    }
    #pragma unroll
    for (int s = 0; s < 4; ++s) {
        int col = tn * 64 + s * 16 + lo;
        float bv = bias[col];
        #pragma unroll
        for (int r = 0; r < 8; ++r) {
            int row = tm * 16 + hi * 8 + r;
            if (row < M) C[(size_t)row * Nout + col] = acc[s][r] + bv;
        }
    }
}

// ---------------- repack q: qkv fp32 [N][3*DIM] -> qb bf16 [NH][NP][HD] (zero padded) ----------------
__global__ void k_extract_q(const float* __restrict__ qkv, __bf16* __restrict__ qb) {
    size_t i = (size_t)blockIdx.x * blockDim.x + threadIdx.x;
    if (i >= (size_t)NP * DIM) return;
    int n = (int)(i / DIM), c = (int)(i % DIM);
    int h = c / HD, d = c % HD;
    float v = (n < N_) ? qkv[(size_t)n * (3 * DIM) + c] : 0.f;
    qb[((size_t)h * NP + n) * HD + d] = (__bf16)v;
}

// ---------------- cls passthrough into pooled buffer ----------------
__global__ void k_pool_cls(const float* __restrict__ qkv, int part_off, float* __restrict__ outp) {
    int i = blockIdx.x * blockDim.x + threadIdx.x;
    if (i >= NH * HD) return;
    int h = i / HD, d = i % HD;
    outp[((size_t)h * KN) * HD + d] = qkv[(size_t)0 * (3 * DIM) + part_off + h * HD + d];
}

// ---------------- depthwise 3x3x3 conv, stride (1,2,2), pad 1 ----------------
__global__ void k_pool_conv(const float* __restrict__ qkv, int part_off,
                            const float* __restrict__ ker, float* __restrict__ outp) {
    size_t i = (size_t)blockIdx.x * blockDim.x + threadIdx.x;
    if (i >= (size_t)NH * KSP * HD) return;
    int d = (int)(i % HD);
    int rest = (int)(i / HD);
    int sp = rest % KSP, h = rest / KSP;
    int ow = sp % KW, oh = (sp / KW) % KH, ot = sp / (KH * KW);
    float s = 0.f;
    #pragma unroll
    for (int dt = -1; dt <= 1; ++dt) {
        int it = ot + dt;
        if (it < 0 || it >= T_) continue;
        #pragma unroll
        for (int dy = -1; dy <= 1; ++dy) {
            int iy = oh * 2 + dy;
            if (iy < 0 || iy >= H_) continue;
            #pragma unroll
            for (int dx = -1; dx <= 1; ++dx) {
                int ix = ow * 2 + dx;
                if (ix < 0 || ix >= W_) continue;
                int tok = 1 + (it * H_ + iy) * W_ + ix;
                float xin = qkv[(size_t)tok * (3 * DIM) + part_off + h * HD + d];
                float kv  = ker[d * 27 + (dt + 1) * 9 + (dy + 1) * 3 + (dx + 1)];
                s += xin * kv;
            }
        }
    }
    outp[((size_t)h * KN + 1 + sp) * HD + d] = s;
}

// ---------------- LayerNorm over HD=64, fp32 in -> bf16 out, 1 wave per row ----------------
// transposed==0: out[h][tok][d] row-major (for K, rows padded to KNP with zeros)
// transposed==1: out[h][d][tok] (V^T layout for PV B-fragments, cols padded to KNP with zeros)
__global__ void k_layernorm_bf16(const float* __restrict__ inp, const float* __restrict__ g,
                                 const float* __restrict__ b, __bf16* __restrict__ outp,
                                 int transposed) {
    int blk = blockIdx.x;                    // h*KNP + tok
    int h = blk / KNP, tok = blk % KNP;
    int l = threadIdx.x;                     // 0..31
    if (tok >= KN) {
        if (!transposed) {
            outp[((size_t)h * KNP + tok) * HD + l]      = (__bf16)0.f;
            outp[((size_t)h * KNP + tok) * HD + l + 32] = (__bf16)0.f;
        } else {
            outp[((size_t)h * HD + l) * KNP + tok]        = (__bf16)0.f;
            outp[((size_t)h * HD + l + 32) * KNP + tok]   = (__bf16)0.f;
        }
        return;
    }
    const float* p = inp + ((size_t)h * KN + tok) * HD;
    float a0 = p[l], a1 = p[l + 32];
    float s = a0 + a1;
    #pragma unroll
    for (int m = 16; m >= 1; m >>= 1) s += __shfl_xor(s, m, 32);
    float mu = s * (1.f / 64.f);
    float d0 = a0 - mu, d1 = a1 - mu;
    float v = d0 * d0 + d1 * d1;
    #pragma unroll
    for (int m = 16; m >= 1; m >>= 1) v += __shfl_xor(v, m, 32);
    float rs = rsqrtf(v * (1.f / 64.f) + 1e-5f);
    float o0 = d0 * rs * g[l] + b[l];
    float o1 = d1 * rs * g[l + 32] + b[l + 32];
    if (!transposed) {
        outp[((size_t)h * KNP + tok) * HD + l]      = (__bf16)o0;
        outp[((size_t)h * KNP + tok) * HD + l + 32] = (__bf16)o1;
    } else {
        outp[((size_t)h * HD + l) * KNP + tok]      = (__bf16)o0;
        outp[((size_t)h * HD + l + 32) * KNP + tok] = (__bf16)o1;
    }
}

// ---------------- decomposed rel-pos tables: out[h][sp][k2] = q . Rp[idx] ----------------
__global__ void k_relpos(const float* __restrict__ qkv, const float* __restrict__ rp,
                         float* __restrict__ out, int kn, int mode, int qn) {
    size_t i = (size_t)blockIdx.x * blockDim.x + threadIdx.x;
    if (i >= (size_t)NH * SP * kn) return;
    int k2 = (int)(i % kn);
    int rest = (int)(i / kn);
    int sp = rest % SP, h = rest / SP;
    int coord;
    if (mode == 0)      coord = sp / (H_ * W_);
    else if (mode == 1) coord = (sp / W_) % H_;
    else                coord = sp % W_;
    int kr = qn / kn;                              // 1 or 2 (q_ratio==1 for these shapes)
    int ri = coord - k2 * kr + (kn - 1) * kr;
    const float* q = qkv + (size_t)(1 + sp) * (3 * DIM) + h * HD;
    const float* r = rp + (size_t)ri * HD;
    float s = 0.f;
    #pragma unroll 8
    for (int c = 0; c < HD; ++c) s += q[c] * r[c];
    out[((size_t)h * SP + sp) * kn + k2] = s;
}

// ---------------- flash attention: 1 wave per (head, 16-row q tile) ----------------
// Q: [h][NP][64] bf16 (padded)   K: [h][KNP][64] bf16 (zero padded)
// Vt: [h][64][KNP] bf16 (zero padded)   out: [NP][DIM] bf16 (padded, unconditional stores)
__global__ void __launch_bounds__(32)
k_flash_attn(const __bf16* __restrict__ qb, const __bf16* __restrict__ kb,
             const __bf16* __restrict__ vt, const float* __restrict__ relh,
             const float* __restrict__ relw, const float* __restrict__ relt,
             __bf16* __restrict__ outb) {
    __shared__ __attribute__((aligned(16))) __bf16 lds_p[16 * 32];

    int h  = blockIdx.x / QT;
    int qt = blockIdx.x % QT;
    int lane = threadIdx.x;
    int lo = lane & 15, hi = lane >> 4;

    const __bf16* Q  = qb + (size_t)h * NP * HD;
    const __bf16* K  = kb + (size_t)h * KNP * HD;
    const __bf16* Vt = vt + (size_t)h * HD * KNP;
    const float* rT = relt + (size_t)h * SP * KT;
    const float* rH = relh + (size_t)h * SP * KH;
    const float* rW = relw + (size_t)h * SP * KW;

    // Q A-fragments (two 32-wide d chunks), unconditional b128 loads
    int qrow = qt * 16 + lo;                       // < NP
    v16bf qa[2];
    #pragma unroll
    for (int c = 0; c < 2; ++c) {
        const __bf16* qp = Q + (size_t)qrow * HD + c * 32 + hi * 8;
        qa[c] = CAT16(*(const v8bf*)qp, *(const v8bf*)(qp + 16));
    }

    // per-row bias-lookup base (clamped spatial index, branchless)
    int rowbase = qt * 16 + hi * 8;
    float mrow[8], lrow[8];
    #pragma unroll
    for (int r = 0; r < 8; ++r) { mrow[r] = -1e30f; lrow[r] = 0.f; }
    v8f oacc[4] = {};

    for (int kb0 = 0; kb0 < KN; kb0 += 32) {
        float pchunk[2][8];
        float chunkmax[8];
        #pragma unroll
        for (int r = 0; r < 8; ++r) chunkmax[r] = -1e30f;

        // ---- two 16x16 S tiles (32 keys) ----
        #pragma unroll
        for (int t = 0; t < 2; ++t) {
            int key = kb0 + t * 16 + lo;           // < KNP (zero-padded rows)
            v8f s = {};
            #pragma unroll
            for (int c = 0; c < 2; ++c) {
                const __bf16* kp = K + (size_t)key * HD + c * 32 + hi * 8;
                v16bf bk = CAT16(*(const v8bf*)kp, *(const v8bf*)(kp + 16));
                s = __builtin_amdgcn_wmma_f32_16x16x32_bf16(false, qa[c], false, bk, (short)0, s, false, false);
            }
            bool kvalid = key < KN;
            bool kcls   = (key == 0);
            int ks  = key - 1; if (ks < 0) ks = 0; if (ks >= KSP) ks = KSP - 1;
            int kw2 = ks % KW, kh2 = (ks / KW) % KH, kt2 = ks / (KH * KW);
            #pragma unroll
            for (int r = 0; r < 8; ++r) {
                int row = rowbase + r;
                bool rv = row < N_;
                int sp = row - 1; if (sp < 0) sp = 0; if (sp >= SP) sp = SP - 1;
                // unconditional table loads, branchless select
                float bsum = rT[(size_t)sp * KT + kt2]
                           + rH[(size_t)sp * KH + kh2]
                           + rW[(size_t)sp * KW + kw2];
                bool usebias = (row >= 1) && !kcls;
                float sv = s[r] * SCALE_ + (usebias ? bsum : 0.f);
                sv = (kvalid && rv) ? sv : -1e30f;
                pchunk[t][r] = sv;
                chunkmax[r] = fmaxf(chunkmax[r], sv);
            }
        }

        // ---- online softmax (row = cross-lane over 16-lane half, per C-layout) ----
        #pragma unroll
        for (int r = 0; r < 8; ++r) {
            float cm = chunkmax[r];
            #pragma unroll
            for (int m = 1; m < 16; m <<= 1) cm = fmaxf(cm, __shfl_xor(cm, m, 32));
            float newm = fmaxf(mrow[r], cm);
            float alpha = __expf(mrow[r] - newm);
            mrow[r] = newm;
            lrow[r] *= alpha;
            #pragma unroll
            for (int dtl = 0; dtl < 4; ++dtl) oacc[dtl][r] *= alpha;
            float p0 = __expf(pchunk[0][r] - newm);
            float p1 = __expf(pchunk[1][r] - newm);
            pchunk[0][r] = p0; pchunk[1][r] = p1;
            float psum = p0 + p1;
            #pragma unroll
            for (int m = 1; m < 16; m <<= 1) psum += __shfl_xor(psum, m, 32);
            lrow[r] += psum;
        }

        // ---- P: C-layout regs -> LDS -> A-fragment layout ----
        #pragma unroll
        for (int t = 0; t < 2; ++t)
            #pragma unroll
            for (int r = 0; r < 8; ++r)
                lds_p[(hi * 8 + r) * 32 + t * 16 + lo] = (__bf16)pchunk[t][r];
        __syncthreads();

        v16bf pa;
        {
            const __bf16* pp = &lds_p[lo * 32 + hi * 8];
            pa = CAT16(*(const v8bf*)pp, *(const v8bf*)(pp + 16));
        }

        // ---- O += P x V via V^T: contiguous b128 loads per lane ----
        #pragma unroll
        for (int dtl = 0; dtl < 4; ++dtl) {
            int dcol = dtl * 16 + lo;
            const __bf16* vp = Vt + (size_t)dcol * KNP + kb0 + hi * 8;
            v16bf bv = CAT16(*(const v8bf*)vp, *(const v8bf*)(vp + 16));
            oacc[dtl] = __builtin_amdgcn_wmma_f32_16x16x32_bf16(false, pa, false, bv, (short)0, oacc[dtl], false, false);
        }
        __syncthreads();
    }

    // ---- epilogue: normalize, write bf16 [NP][DIM] (padded -> unconditional) ----
    #pragma unroll
    for (int r = 0; r < 8; ++r) {
        int row = rowbase + r;
        float inv = 1.f / lrow[r];
        #pragma unroll
        for (int dtl = 0; dtl < 4; ++dtl) {
            int dcol = dtl * 16 + lo;
            outb[(size_t)row * DIM + h * HD + dcol] = (__bf16)(oacc[dtl][r] * inv);
        }
    }
}

// ---------------- host launch ----------------
static inline int cdiv(long a, long b) { return (int)((a + b - 1) / b); }

extern "C" void kernel_launch(void* const* d_in, const int* in_sizes, int n_in,
                              void* d_out, int out_size, void* d_ws, size_t ws_size,
                              hipStream_t stream) {
    const float* x      = (const float*)d_in[0];
    const float* qkv_w  = (const float*)d_in[1];
    const float* qkv_b  = (const float*)d_in[2];
    const float* proj_w = (const float*)d_in[3];
    const float* proj_b = (const float*)d_in[4];
    const float* pkw    = (const float*)d_in[5];
    const float* pvw    = (const float*)d_in[6];
    const float* nkg    = (const float*)d_in[7];
    const float* nkb    = (const float*)d_in[8];
    const float* nvg    = (const float*)d_in[9];
    const float* nvb    = (const float*)d_in[10];
    const float* rph    = (const float*)d_in[11];
    const float* rpw    = (const float*)d_in[12];
    const float* rpt    = (const float*)d_in[13];

    // carve workspace (~118 MB total)
    char* p = (char*)d_ws;
    auto carve = [&](size_t bytes) -> void* {
        void* r = (void*)p;
        p += (bytes + 255) & ~(size_t)255;
        return r;
    };
    __bf16* xb   = (__bf16*)carve((size_t)N_ * DIM * 2);
    __bf16* wqb  = (__bf16*)carve((size_t)3 * DIM * DIM * 2);
    __bf16* wpb  = (__bf16*)carve((size_t)DIM * DIM * 2);
    float*  qkvf = (float*) carve((size_t)N_ * 3 * DIM * 4);
    float*  kpf  = (float*) carve((size_t)NH * KN * HD * 4);
    float*  vpf  = (float*) carve((size_t)NH * KN * HD * 4);
    __bf16* qb   = (__bf16*)carve((size_t)NH * NP * HD * 2);
    __bf16* kbb  = (__bf16*)carve((size_t)NH * KNP * HD * 2);
    __bf16* vtb  = (__bf16*)carve((size_t)NH * HD * KNP * 2);
    float*  relh = (float*) carve((size_t)NH * SP * KH * 4);
    float*  relw = (float*) carve((size_t)NH * SP * KW * 4);
    float*  relt = (float*) carve((size_t)NH * SP * KT * 4);
    __bf16* aob  = (__bf16*)carve((size_t)NP * DIM * 2);

    const int TB = 256;

    // 1) fp32 -> bf16 for GEMM operands
    k_f32_to_bf16<<<cdiv((long)N_ * DIM, TB), TB, 0, stream>>>(x, xb, (size_t)N_ * DIM);
    k_f32_to_bf16<<<cdiv((long)3 * DIM * DIM, TB), TB, 0, stream>>>(qkv_w, wqb, (size_t)3 * DIM * DIM);
    k_f32_to_bf16<<<cdiv((long)DIM * DIM, TB), TB, 0, stream>>>(proj_w, wpb, (size_t)DIM * DIM);

    // 2) QKV GEMM (WMMA): qkvf[N][2304] = x @ qkv_w^T + qkv_b
    {
        long waves = (long)((N_ + 15) / 16) * (3 * DIM / 64);
        k_gemm_bf16<<<cdiv(waves * 32, TB), TB, 0, stream>>>(xb, wqb, qkv_b, qkvf, N_, DIM, 3 * DIM);
    }

    // 3) repack q to [NH][NP][HD] bf16 (zero padded rows)
    k_extract_q<<<cdiv((long)NP * DIM, TB), TB, 0, stream>>>(qkvf, qb);

    // 4) pool k,v: cls passthrough + depthwise conv
    k_pool_cls<<<cdiv(NH * HD, TB), TB, 0, stream>>>(qkvf, DIM, kpf);
    k_pool_cls<<<cdiv(NH * HD, TB), TB, 0, stream>>>(qkvf, 2 * DIM, vpf);
    k_pool_conv<<<cdiv((long)NH * KSP * HD, TB), TB, 0, stream>>>(qkvf, DIM, pkw, kpf);
    k_pool_conv<<<cdiv((long)NH * KSP * HD, TB), TB, 0, stream>>>(qkvf, 2 * DIM, pvw, vpf);

    // 5) layernorm -> bf16 K (row-major, padded) and V^T (transposed, padded)
    k_layernorm_bf16<<<NH * KNP, 32, 0, stream>>>(kpf, nkg, nkb, kbb, 0);
    k_layernorm_bf16<<<NH * KNP, 32, 0, stream>>>(vpf, nvg, nvb, vtb, 1);

    // 6) decomposed rel-pos bias tables
    k_relpos<<<cdiv((long)NH * SP * KT, TB), TB, 0, stream>>>(qkvf, rpt, relt, KT, 0, T_);
    k_relpos<<<cdiv((long)NH * SP * KH, TB), TB, 0, stream>>>(qkvf, rph, relh, KH, 1, H_);
    k_relpos<<<cdiv((long)NH * SP * KW, TB), TB, 0, stream>>>(qkvf, rpw, relw, KW, 2, W_);

    // 7) flash attention (WMMA QK^T + PV, online softmax, bias on the fly)
    k_flash_attn<<<NH * QT, 32, 0, stream>>>(qb, kbb, vtb, relh, relw, relt, aob);

    // 8) output projection (WMMA): d_out[N][768] = aob @ proj_w^T + proj_b
    {
        long waves = (long)((N_ + 15) / 16) * (DIM / 64);
        k_gemm_bf16<<<cdiv(waves * 32, TB), TB, 0, stream>>>(aob, wpb, proj_b, (float*)d_out, N_, DIM, DIM);
    }
}
